// Head_72146860638778
// MI455X (gfx1250) — compile-verified, matrix operand order checked by reading, not run on hardware
//
#include <hip/hip_runtime.h>

// Problem constants (match reference)
static constexpr int BB = 16;     // batch
static constexpr int TT = 2048;   // sequence
static constexpr int CC = 1024;   // model dim
static constexpr int HH = 256;    // head dim
#define INV_SCALE 0.022097086912079608f   // 1/sqrt(2048)

typedef __attribute__((ext_vector_type(16))) __bf16 v16bf;
typedef __attribute__((ext_vector_type(2)))  __bf16 v2bf;
typedef __attribute__((ext_vector_type(8)))  float  v8f;
typedef __attribute__((ext_vector_type(8)))  unsigned int v8u;
typedef __attribute__((ext_vector_type(4)))  unsigned int u32x4;
typedef __attribute__((ext_vector_type(4)))  float  f32x4;
typedef __attribute__((ext_vector_type(8)))  int    i32x8;
typedef __attribute__((ext_vector_type(4)))  int    i32x4;

#if defined(__HIP_DEVICE_COMPILE__) && __has_builtin(__builtin_amdgcn_tensor_load_to_lds) && __has_builtin(__builtin_amdgcn_s_wait_tensorcnt)
#define USE_TDM 1
#else
#define USE_TDM 0
#endif

__device__ __forceinline__ unsigned short f2bf(float f) {
  unsigned int u = __builtin_bit_cast(unsigned int, f);
  u += 0x7FFFu + ((u >> 16) & 1u);           // round-to-nearest-even
  return (unsigned short)(u >> 16);
}
__device__ __forceinline__ unsigned pack2bf(float a, float b) {
#if defined(__HIP_DEVICE_COMPILE__) && __has_builtin(__builtin_amdgcn_cvt_pk_bf16_f32)
  v2bf t = __builtin_amdgcn_cvt_pk_bf16_f32(a, b);
  return __builtin_bit_cast(unsigned, t);
#else
  return (unsigned)f2bf(a) | ((unsigned)f2bf(b) << 16);
#endif
}

// 16x32 bf16 WMMA A-fragment (or B-fragment when the matrix is stored
// column-contiguous-in-K) from LDS; `ld` in bf16 elements, must be even.
__device__ __forceinline__ v16bf frag_ld(const unsigned short* p0, int ld, int lane) {
  const int half = (lane >> 4) & 1;
  const int r = lane & 15;
  const unsigned int* p = (const unsigned int*)(p0 + r * ld);
  v8u au;
#pragma unroll
  for (int v = 0; v < 8; ++v) {
    const int k0 = ((v < 4) ? (v << 1) : (16 + ((v - 4) << 1))) + (half << 3);
    au[v] = p[k0 >> 1];
  }
  return __builtin_bit_cast(v16bf, au);
}

#if USE_TDM
__device__ __forceinline__ unsigned lds_off(const void* p) {
  return (unsigned)(unsigned long long)(__attribute__((address_space(3))) const void*)p;
}
// TDM 2D tile load: `rows` lines of 256 bf16 (512B), row stride 256 elements in
// global memory; LDS destination padded +1 dword per 128-dword line => LDS row
// stride 258 bf16 elements (matches frag_ld layout above).
__device__ __forceinline__ void tdm_load_rows256(unsigned lds_byte,
                                                 const unsigned short* gptr,
                                                 int rows) {
  unsigned long long ga = (unsigned long long)gptr;
  u32x4 g0;
  g0[0] = 1u;                                          // count=1, user mode
  g0[1] = lds_byte;                                    // LDS dest
  g0[2] = (unsigned)(ga & 0xFFFFFFFFu);                // global addr lo
  g0[3] = (unsigned)((ga >> 32) & 0x01FFFFFFu) | (2u << 30);  // hi + type=2
  i32x8 g1;
  g1[0] = (1 << 16)     // data_size = 2B
        | (1 << 20)     // pad_enable
        | (6 << 22);    // pad_interval: 128 dwords; pad_amount[31:25]=0 => 1 dword
  g1[1] = (int)(256u << 16);                    // tensor_dim0 = 256 (bits 79:48)
  g1[2] = (int)(((unsigned)rows) << 16);        // tensor_dim1 lo (bits 111:80)
  g1[3] = (int)(256u << 16);                    // tile_dim0 = 256 (bits 127:112)
  g1[4] = rows & 0xFFFF;                        // tile_dim1, tile_dim2=0 (2D)
  g1[5] = 256;                                  // tensor_dim0_stride lo32
  g1[6] = (int)(256u << 16);                    // dim0_stride hi=0 | dim1_stride lo16
  g1[7] = 0;                                    // dim1_stride hi32
  i32x4 gz4 = {0, 0, 0, 0};
  i32x8 gz8 = {0, 0, 0, 0, 0, 0, 0, 0};
  __builtin_amdgcn_tensor_load_to_lds(g0, g1, gz4, gz4, gz8, 0);
}
#endif

// ---------------------------------------------------------------------------
// Kernel 1: fused QKV projection. Computes bf16(x@Wq+bq), (..Wk..), (..Wv..)
// in ONE workgroup per (64-row, 64-col) tile so the x tile is staged once and
// reused by all three weight matrices. Grid: (M/64, H/64), block 256 (8 waves).
// ---------------------------------------------------------------------------
__global__ __launch_bounds__(256) void qkv_proj(
    const float* __restrict__ x,
    const float* __restrict__ Wq, const float* __restrict__ bq,
    const float* __restrict__ Wk, const float* __restrict__ bk,
    const float* __restrict__ Wv, const float* __restrict__ bv,
    unsigned short* __restrict__ Qw, unsigned short* __restrict__ Kw,
    unsigned short* __restrict__ Vw) {
  __shared__ __align__(16) unsigned short xs[64 * 34];       // [row][k]
  __shared__ __align__(16) unsigned short wt[3][64 * 34];    // [n][k] (W^T)

  const int tid  = threadIdx.x;
  const int lane = tid & 31;
  const int w    = tid >> 5;
  const int mi   = w & 3;
  const int ni   = w >> 2;
  const int rowBase = blockIdx.x * 64;
  const int nBase   = blockIdx.y * 64;

  const float* Ws[3] = {Wq, Wk, Wv};
  const float* bs[3] = {bq, bk, bv};
  unsigned short* outs[3] = {Qw, Kw, Vw};

  v8f zf = {0.f, 0.f, 0.f, 0.f, 0.f, 0.f, 0.f, 0.f};
  v8f acc[3][2] = {{zf, zf}, {zf, zf}, {zf, zf}};

  for (int kB = 0; kB < CC; kB += 32) {
    __syncthreads();
    // stage x tile: 64x32 f32 -> bf16; 512 float4 chunks, 2 per thread
#pragma unroll
    for (int i = 0; i < 2; ++i) {
      int c4 = tid + i * 256;
      int row = c4 >> 3, kc = (c4 & 7) * 4;
      f32x4 f = *(const f32x4*)(x + (size_t)(rowBase + row) * CC + kB + kc);
      unsigned* dst = (unsigned*)(xs + row * 34 + kc);
      dst[0] = pack2bf(f[0], f[1]);
      dst[1] = pack2bf(f[2], f[3]);
    }
    // stage W^T tiles for all three matrices: 32x64 f32 each -> bf16 [n][k]
#pragma unroll
    for (int z = 0; z < 3; ++z)
#pragma unroll
      for (int i = 0; i < 2; ++i) {
        int c4 = tid + i * 256;
        int n4 = (c4 & 15) * 4, kk = c4 >> 4;
        f32x4 f = *(const f32x4*)(Ws[z] + (size_t)(kB + kk) * HH + nBase + n4);
#pragma unroll
        for (int j = 0; j < 4; ++j) wt[z][(n4 + j) * 34 + kk] = f2bf(f[j]);
      }
    __syncthreads();
    v16bf a = frag_ld(xs + mi * 16 * 34, 34, lane);
#pragma unroll
    for (int z = 0; z < 3; ++z) {
      v16bf b0 = frag_ld(wt[z] + (ni * 32) * 34, 34, lane);
      v16bf b1 = frag_ld(wt[z] + (ni * 32 + 16) * 34, 34, lane);
      acc[z][0] = __builtin_amdgcn_wmma_f32_16x16x32_bf16(false, a, false, b0, (short)0, acc[z][0], false, false);
      acc[z][1] = __builtin_amdgcn_wmma_f32_16x16x32_bf16(false, a, false, b1, (short)0, acc[z][1], false, false);
    }
  }

  const int half = lane >> 4;
  const int n = lane & 15;
#pragma unroll
  for (int z = 0; z < 3; ++z) {
    const float bias0 = bs[z][nBase + ni * 32 + n];
    const float bias1 = bs[z][nBase + ni * 32 + 16 + n];
#pragma unroll
    for (int r = 0; r < 8; ++r) {
      int row = rowBase + mi * 16 + r + half * 8;
      size_t o = (size_t)row * HH + nBase + ni * 32;
      outs[z][o + n]      = f2bf(acc[z][0][r] + bias0);
      outs[z][o + 16 + n] = f2bf(acc[z][1][r] + bias1);
    }
  }
}

// ---------------------------------------------------------------------------
// Kernel 2: flash attention (online softmax). Grid (T/32, B), block 256.
// Q tile = 32 rows, kv block = 32 (causal: nkv = blockIdx.x + 1).
// Q fragments are preloaded into registers, after which the Q LDS buffer is
// reused as the second K buffer: K tiles are double-buffered via TDM so the
// DMA of block jb+1 overlaps softmax + P@V of block jb.
// ---------------------------------------------------------------------------
__global__ __launch_bounds__(256) void flash_attn(
    const unsigned short* __restrict__ Qw,
    const unsigned short* __restrict__ Kw,
    const unsigned short* __restrict__ Vw,
    float* __restrict__ out) {
  __shared__ __align__(16) unsigned short Qs[32 * 258];  // Q, then K buf 1
  __shared__ __align__(16) unsigned short Ks[32 * 258];  // K buf 0
  __shared__ __align__(16) unsigned short Vt[256 * 34];  // [h][kv]
  __shared__ __align__(16) float          Sb[32 * 33];   // scores f32
  __shared__ __align__(16) unsigned short Pb[32 * 34];   // probs bf16
  __shared__ float m_s[32], l_s[32], a_s[32];

  const int tid  = threadIdx.x;
  const int lane = tid & 31;
  const int w    = tid >> 5;
  const int qBase = blockIdx.x * 32;
  const size_t boff = (size_t)blockIdx.y * TT * HH;
  const unsigned short* Qb = Qw + boff;
  const unsigned short* Kb = Kw + boff;
  const unsigned short* Vb = Vw + boff;
  const int nkv = blockIdx.x + 1;

  if (tid < 32) { m_s[tid] = -3.0e38f; l_s[tid] = 0.f; }

  // ---- stage Q (32x256) and kick off K block 0 ----
#if USE_TDM
  if (w == 0) {
    tdm_load_rows256(lds_off(Qs), Qb + (size_t)qBase * HH, 32);
    tdm_load_rows256(lds_off(Ks), Kb, 32);               // K block 0
    __builtin_amdgcn_s_wait_tensorcnt(1);                // in-order: Q complete
  }
#else
#pragma unroll
  for (int i = 0; i < 4; ++i) {
    int c8 = tid + i * 256;
    int qi = c8 >> 5, h8 = (c8 & 31) * 8;
    u32x4 u = *(const u32x4*)(Qb + (size_t)(qBase + qi) * HH + h8);
    unsigned* dst = (unsigned*)(Qs + qi * 258 + h8);
    dst[0] = u[0]; dst[1] = u[1]; dst[2] = u[2]; dst[3] = u[3];
  }
#endif
  __syncthreads();                                       // Q visible

  // ---- preload loop-invariant Q fragments into registers (waves 0-3) ----
  v16bf qf[8];
  if (w < 4) {
    const int smi = w & 1;
#pragma unroll
    for (int kk = 0; kk < 8; ++kk)
      qf[kk] = frag_ld(Qs + smi * 16 * 258 + kk * 32, 258, lane);
  }

  v8f zf = {0.f, 0.f, 0.f, 0.f, 0.f, 0.f, 0.f, 0.f};
  v8f oAcc[2][2] = {{zf, zf}, {zf, zf}};                 // [Msub][Nsub] 16x16 f32

  for (int jb = 0; jb < nkv; ++jb) {
    const int kvBase = jb * 32;
    __syncthreads();   // prior compute done; also: Q preload done before K1 DMA
    // ---- stage V transposed (manual); K handled by TDM ping-pong ----
#pragma unroll
    for (int i = 0; i < 4; ++i) {
      int c8 = tid + i * 256;
      int s = c8 >> 5, h8 = (c8 & 31) * 8;
      u32x4 uv = *(const u32x4*)(Vb + (size_t)(kvBase + s) * HH + h8);
#pragma unroll
      for (int j = 0; j < 4; ++j) {
        Vt[(h8 + 2 * j) * 34 + s]     = (unsigned short)(uv[j] & 0xFFFFu);
        Vt[(h8 + 2 * j + 1) * 34 + s] = (unsigned short)(uv[j] >> 16);
      }
#if !USE_TDM
      u32x4 uk = *(const u32x4*)(Kb + (size_t)(kvBase + s) * HH + h8);
      unsigned* dk = (unsigned*)(Ks + s * 258 + h8);
      dk[0] = uk[0]; dk[1] = uk[1]; dk[2] = uk[2]; dk[3] = uk[3];
#endif
    }
#if USE_TDM
    if (w == 0) {
      __builtin_amdgcn_s_wait_tensorcnt(0);              // K[jb] DMA complete
      if (jb + 1 < nkv)                                  // prefetch K[jb+1]
        tdm_load_rows256((jb & 1) ? lds_off(Ks) : lds_off(Qs),
                         Kb + (size_t)(kvBase + 32) * HH, 32);
    }
    const unsigned short* kbuf = (jb & 1) ? Qs : Ks;
#else
    const unsigned short* kbuf = Ks;
#endif
    __syncthreads();                                     // V + K[jb] visible

    if (w < 4) {                                         // S = Q @ K^T (32x32)
      const int smi = w & 1, ks = w >> 1;
      v8f sa = zf;
#pragma unroll
      for (int kk = 0; kk < 8; ++kk) {
        v16bf bm = frag_ld(kbuf + ks * 16 * 258 + kk * 32, 258, lane);
        sa = __builtin_amdgcn_wmma_f32_16x16x32_bf16(false, qf[kk], false, bm, (short)0, sa, false, false);
      }
      const int half = lane >> 4, n = lane & 15;
#pragma unroll
      for (int r = 0; r < 8; ++r) {
        int m  = smi * 16 + r + half * 8;
        int kc = ks * 16 + n;
        float v = sa[r] * INV_SCALE;
        if (kvBase + kc > qBase + m) v = -3.0e38f;       // causal mask
        Sb[m * 33 + kc] = v;
      }
    }
    __syncthreads();

    if (tid < 32) {                                      // online softmax
      const int q = tid;
      float mo = m_s[q];
      float rm = mo;
      for (int j = 0; j < 32; ++j) rm = fmaxf(rm, Sb[q * 33 + j]);
      float alpha = __expf(mo - rm);
      float l = l_s[q] * alpha;
      for (int j = 0; j < 32; ++j) {
        float p = __expf(Sb[q * 33 + j] - rm);
        l += p;
        Pb[q * 34 + j] = f2bf(p);
      }
      m_s[q] = rm; l_s[q] = l; a_s[q] = alpha;
    }
    __syncthreads();

    {                                                    // O = O*alpha + P @ V
      const int half = lane >> 4;
#pragma unroll
      for (int mi = 0; mi < 2; ++mi) {
        float al[8];
#pragma unroll
        for (int r = 0; r < 8; ++r) al[r] = a_s[mi * 16 + r + half * 8];
#pragma unroll
        for (int nt = 0; nt < 2; ++nt)
#pragma unroll
          for (int r = 0; r < 8; ++r) oAcc[mi][nt][r] *= al[r];
        v16bf a = frag_ld(Pb + mi * 16 * 34, 34, lane);
#pragma unroll
        for (int nt = 0; nt < 2; ++nt) {
          v16bf bm = frag_ld(Vt + (w * 32 + nt * 16) * 34, 34, lane);
          oAcc[mi][nt] = __builtin_amdgcn_wmma_f32_16x16x32_bf16(false, a, false, bm, (short)0, oAcc[mi][nt], false, false);
        }
      }
    }
  }

  const int half = lane >> 4, n = lane & 15;
#pragma unroll
  for (int mi = 0; mi < 2; ++mi)
#pragma unroll
    for (int nt = 0; nt < 2; ++nt)
#pragma unroll
      for (int r = 0; r < 8; ++r) {
        int row = mi * 16 + r + half * 8;
        float inv = 1.0f / l_s[row];
        out[boff + (size_t)(qBase + row) * HH + w * 32 + nt * 16 + n] = oAcc[mi][nt][r] * inv;
      }
}

extern "C" void kernel_launch(void* const* d_in, const int* in_sizes, int n_in,
                              void* d_out, int out_size, void* d_ws, size_t ws_size,
                              hipStream_t stream) {
  (void)in_sizes; (void)n_in; (void)out_size; (void)ws_size;
  const float* x  = (const float*)d_in[0];
  const float* Wq = (const float*)d_in[1];
  const float* bq = (const float*)d_in[2];
  const float* Wk = (const float*)d_in[3];
  const float* bk = (const float*)d_in[4];
  const float* Wv = (const float*)d_in[5];
  const float* bv = (const float*)d_in[6];

  // bf16 Q/K/V in workspace: 3 * 16*2048*256 * 2B = 48 MiB
  const size_t BTH = (size_t)BB * TT * HH;
  unsigned short* Qw = (unsigned short*)d_ws;
  unsigned short* Kw = Qw + BTH;
  unsigned short* Vw = Kw + BTH;

  dim3 gProj((BB * TT) / 64, HH / 64);
  qkv_proj<<<gProj, 256, 0, stream>>>(x, Wq, bq, Wk, bk, Wv, bv, Qw, Kw, Vw);

  dim3 gAttn(TT / 32, BB);
  flash_attn<<<gAttn, 256, 0, stream>>>(Qw, Kw, Vw, (float*)d_out);
}
